// MoEFFN_90323162235698
// MI455X (gfx1250) — matrix operand order repaired and checked by reading, not compiled, not run on hardware
//
#include <hip/hip_runtime.h>
#include <hip/hip_bf16.h>

#define NTOK   8192
#define DMODEL 1024
#define NEXP   8
#define HDIM   2816
#define RBLK   32
#define HCHUNK 64
#define KSTG   128    // staged K-slice depth for G/U weights
#define XLD    1032   // padded f16 row stride for X tile (16B-aligned rows, conflict-free)
#define KPAD   136    // padded K stride for transposed G/U slices (multiple of 8 halfs)
#define KPAD2  40     // padded K stride for transposed Down slice (multiple of 8 halfs)

typedef __attribute__((ext_vector_type(16))) _Float16 v16h;
typedef __attribute__((ext_vector_type(8)))  float    v8f;

union Frag16 { v16h v; _Float16 h[16]; unsigned u[8]; uint4 q[2]; };

// ---- A fragment (f16 16x32, ISA 7.12.2): two contiguous 16B runs -> 2x ds_load_b128 ----
__device__ __forceinline__ v16h load_a_lds(const _Float16* base, int lda, int k0, int lane) {
  Frag16 f;
  int hl = lane >> 4, m = lane & 15;
  const _Float16* rowp = base + m * lda + k0 + hl * 8;   // K = k0+8*hl+{0..7}, then +16
  f.q[0] = *(const uint4*)rowp;        // VGPR 0..3 : K pairs 0,2,4,6 (+hl*8)
  f.q[1] = *(const uint4*)(rowp + 16); // VGPR 4..7 : K pairs 16,18,20,22 (+hl*8)
  return f.v;
}

// ---- B fragment (f16 32x16, K = 8*halfwave + {0..7}, then +16) from transposed LDS slice ----
__device__ __forceinline__ v16h load_b_lds(const _Float16* Bs, int ldb, int ks,
                                           int col0, int lane) {
  Frag16 f;
  int hl = lane >> 4, n = lane & 15;
  const _Float16* p = Bs + (col0 + n) * ldb + ks + hl * 8;  // K = ks+8*hl+{0..7}, then +16
  f.q[0] = *(const uint4*)p;
  f.q[1] = *(const uint4*)(p + 16);
  return f.v;
}

__device__ __forceinline__ v8f wmma_f16(v16h a, v16h b, v8f c) {
  return __builtin_amdgcn_wmma_f32_16x16x32_f16(false, a, false, b, (short)0, c, false, false);
}

// pack two f32 rows (k, k+1) of one column into a packed-f16 dword (v_cvt_pk_f16_f32)
__device__ __forceinline__ void pack_pair(_Float16* dst, float lo, float hi) {
  union { unsigned u; _Float16 h[2]; } pk;
  pk.h[0] = (_Float16)lo;
  pk.h[1] = (_Float16)hi;
  *(unsigned*)dst = pk.u;
}

// ================= init =================
__global__ void moe_init_kernel(unsigned* __restrict__ cnt, float* __restrict__ fP) {
  int t = threadIdx.x;
  if (t < NEXP) cnt[t] = 0u;
  if (t < 2 * NEXP) fP[t] = 0.0f;
}

// ================= router: one wave per token =================
__global__ void __launch_bounds__(256) moe_router_kernel(
    const float* __restrict__ xf, const float* __restrict__ gw,
    unsigned* __restrict__ cnt, float* __restrict__ fP,
    int* __restrict__ tok_idx, float* __restrict__ tok_w) {
  __shared__ float sfP[16];
  int tid = threadIdx.x;
  int lane = tid & 31, wv = tid >> 5;
  if (tid < 16) sfP[tid] = 0.0f;
  __syncthreads();

  int n = blockIdx.x * 8 + wv;
  float acc[NEXP];
#pragma unroll
  for (int e = 0; e < NEXP; ++e) acc[e] = 0.0f;
  const float* xrow = xf + (size_t)n * DMODEL;
  for (int d = lane; d < DMODEL; d += 32) {
    float xv = xrow[d];
    const float* g = gw + d * NEXP;
#pragma unroll
    for (int e = 0; e < NEXP; ++e) acc[e] += xv * g[e];
  }
#pragma unroll
  for (int e = 0; e < NEXP; ++e) {
#pragma unroll
    for (int off = 16; off > 0; off >>= 1) acc[e] += __shfl_down(acc[e], off);
  }
  if (lane == 0) {
    float mx = acc[0];
#pragma unroll
    for (int e = 1; e < NEXP; ++e) mx = fmaxf(mx, acc[e]);
    float p[NEXP], se = 0.0f;
#pragma unroll
    for (int e = 0; e < NEXP; ++e) { p[e] = expf(acc[e] - mx); se += p[e]; }
    int i1 = 0;
#pragma unroll
    for (int e = 1; e < NEXP; ++e) if (acc[e] > acc[i1]) i1 = e;
    int i2 = (i1 == 0) ? 1 : 0;
#pragma unroll
    for (int e = 0; e < NEXP; ++e) if (e != i1 && acc[e] > acc[i2]) i2 = e;
    float m2 = fmaxf(acc[i1], acc[i2]);
    float e1 = expf(acc[i1] - m2), e2 = expf(acc[i2] - m2);
    float inv = 1.0f / (e1 + e2);
    atomicAdd(&sfP[i1], 1.0f);                 // f: top-1 one-hot count
#pragma unroll
    for (int e = 0; e < NEXP; ++e) atomicAdd(&sfP[8 + e], p[e] / se);  // P: softmax mean
    unsigned p1 = atomicAdd(&cnt[i1], 1u);
    tok_idx[(size_t)i1 * NTOK + p1] = n; tok_w[(size_t)i1 * NTOK + p1] = e1 * inv;
    unsigned p2 = atomicAdd(&cnt[i2], 1u);
    tok_idx[(size_t)i2 * NTOK + p2] = n; tok_w[(size_t)i2 * NTOK + p2] = e2 * inv;
  }
  __syncthreads();
  if (tid < 16) atomicAdd(&fP[tid], sfP[tid]);
}

// ================= aux loss =================
__global__ void moe_aux_kernel(const float* __restrict__ fP, float* __restrict__ outAux) {
  if (threadIdx.x == 0) {
    const float invN = 1.0f / (float)NTOK;
    float s = 0.0f;
#pragma unroll
    for (int e = 0; e < NEXP; ++e) s += (fP[e] * invN) * (fP[8 + e] * invN);
    outAux[0] = 0.01f * (float)NEXP * s;
  }
}

// ================= fused grok-FFN: down(gelu(xG) * xU) with WMMA =================
__global__ void __launch_bounds__(256) moe_ffn_kernel(
    const float* __restrict__ xf,
    const float* __restrict__ Wg, const float* __restrict__ Wu, const float* __restrict__ Wd,
    const int* __restrict__ tok_idx, const float* __restrict__ tok_w,
    const unsigned* __restrict__ cnt,
    float* __restrict__ out, int shared_pass) {
  extern __shared__ char smem[];
  _Float16* Xs   = (_Float16*)smem;                       // RBLK x XLD (f16)
  _Float16* ActS = Xs + RBLK * XLD;                       // RBLK x HCHUNK
  _Float16* Pool = ActS + RBLK * HCHUNK;                  // time-shared staging pool
  _Float16* Gs   = Pool;                                  // HCHUNK x KPAD  (stage 1)
  _Float16* Us   = Gs + HCHUNK * KPAD;                    // HCHUNK x KPAD  (stage 1)
  _Float16* DnS  = Pool;                                  // DMODEL x KPAD2 (stage 2, aliases)
  float*    Wrow = (float*)(Pool + DMODEL * KPAD2);       // RBLK combine weights
  int*      Trow = (int*)(Wrow + RBLK);                   // RBLK token ids

  int e = blockIdx.y;
  size_t gu_base = shared_pass ? 0 : (size_t)e * DMODEL * HDIM;
  size_t d_base  = shared_pass ? 0 : (size_t)e * HDIM * DMODEL;
  const float* G  = Wg + gu_base;
  const float* U  = Wu + gu_base;
  const float* Dn = Wd + d_base;

  int row0  = blockIdx.x * RBLK;
  int nrows = shared_pass ? NTOK : (int)cnt[e];
  if (row0 >= nrows) return;                              // gathered-MoE sparsity

  int tid = threadIdx.x;
  if (tid < RBLK) {
    int slot = row0 + tid;
    if (shared_pass)        { Trow[tid] = slot; Wrow[tid] = 1.0f; }
    else if (slot < nrows)  { Trow[tid] = tok_idx[(size_t)e * NTOK + slot];
                              Wrow[tid] = tok_w[(size_t)e * NTOK + slot]; }
    else                    { Trow[tid] = 0;    Wrow[tid] = 0.0f; }   // padded row: weight 0
  }
  __syncthreads();

  // stage gathered X rows into LDS as f16
  for (int i = tid; i < RBLK * DMODEL; i += 256) {
    int r = i >> 10, c = i & (DMODEL - 1);
    Xs[r * XLD + c] = (_Float16)xf[(size_t)Trow[r] * DMODEL + c];
  }
  __syncthreads();

  int lane = tid & 31, wv = tid >> 5;
  int rt1 = wv & 1, ct1 = wv >> 1;        // stage-1 tile owned by this wave
  int colbase2 = wv * 128;                // stage-2 output column strip

  const v8f vzero = {0, 0, 0, 0, 0, 0, 0, 0};
  v8f acc[2][8];
#pragma unroll
  for (int a = 0; a < 2; ++a)
#pragma unroll
    for (int b = 0; b < 8; ++b) acc[a][b] = vzero;

  for (int hc = 0; hc < HDIM; hc += HCHUNK) {
    __builtin_prefetch(&Dn[(size_t)hc * DMODEL + colbase2], 0, 0);
    // ---- stage 1: T1 = X*G tile, T2 = X*U tile (16x16 each, same position) ----
    v8f t1 = vzero, t2 = vzero;
    for (int k0 = 0; k0 < DMODEL; k0 += KSTG) {
      __syncthreads();   // previous pool consumers done (G/U slice or Down slice)
      // cooperative stage of G/U (KSTG x HCHUNK) f32 -> transposed, K-pair-packed f16
      // tasks: 2 matrices x (KSTG/2=64 row pairs) x (HCHUNK/4=16 col groups) = 2048
      for (int t = tid; t < 2048; t += 256) {
        int mat = t >> 10;
        int rem = t & 1023;
        int rp  = rem >> 4;       // row pair -> K = k0+2rp, k0+2rp+1
        int cg  = rem & 15;       // 4-wide column group
        const float* src = (mat ? U : G) + (size_t)(k0 + 2 * rp) * HDIM + (hc + cg * 4);
        float4 r0 = *(const float4*)src;          // global_load_b128
        float4 r1 = *(const float4*)(src + HDIM);
        _Float16* dst = (mat ? Us : Gs) + 2 * rp;
        pack_pair(dst + (cg * 4 + 0) * KPAD, r0.x, r1.x);
        pack_pair(dst + (cg * 4 + 1) * KPAD, r0.y, r1.y);
        pack_pair(dst + (cg * 4 + 2) * KPAD, r0.z, r1.z);
        pack_pair(dst + (cg * 4 + 3) * KPAD, r0.w, r1.w);
      }
      __syncthreads();
      // consume the slice: all fragments are 2x ds_load_b128
#pragma unroll
      for (int ks = 0; ks < KSTG; ks += 32) {
        v16h af = load_a_lds(Xs + rt1 * 16 * XLD, XLD, k0 + ks, lane);
        v16h bg = load_b_lds(Gs, KPAD, ks, ct1 * 16, lane);
        v16h bu = load_b_lds(Us, KPAD, ks, ct1 * 16, lane);
        t1 = wmma_f16(af, bg, t1);
        t2 = wmma_f16(af, bu, t2);
      }
    }
    __syncthreads();   // previous chunk's ActS consumers are done
    // exact-erf gelu, elementwise mul (identical C-layouts), write Act tile as f16
#pragma unroll
    for (int j = 0; j < 8; ++j) {
      float g = t1[j];
      float act = 0.5f * g * (1.0f + erff(g * 0.70710678118f)) * t2[j];
      int M    = rt1 * 16 + j + ((lane >> 4) << 3);
      int Ncol = ct1 * 16 + (lane & 15);
      ActS[M * HCHUNK + Ncol] = (_Float16)act;
    }
    __syncthreads();
    // ---- stage 2: OUT strip += Act(32xHCHUNK) @ Down(HCHUNK x 1024) ----
    for (int kk = 0; kk < HCHUNK; kk += 32) {
      __syncthreads();   // previous pool consumers done (Gs/Us dead; prior Down slice read)
      // cooperative stage of Down (32 x DMODEL) f32 -> transposed, K-pair-packed f16
      // tasks: (16 row pairs) x (DMODEL/4=256 col groups) = 4096
      for (int t = tid; t < 4096; t += 256) {
        int rp = t >> 8;          // row pair -> K = hc+kk+2rp, +1
        int cg = t & 255;         // 4-wide column group
        const float* src = Dn + (size_t)(hc + kk + 2 * rp) * DMODEL + cg * 4;
        float4 r0 = *(const float4*)src;          // global_load_b128
        float4 r1 = *(const float4*)(src + DMODEL);
        _Float16* dst = DnS + 2 * rp;
        pack_pair(dst + (cg * 4 + 0) * KPAD2, r0.x, r1.x);
        pack_pair(dst + (cg * 4 + 1) * KPAD2, r0.y, r1.y);
        pack_pair(dst + (cg * 4 + 2) * KPAD2, r0.z, r1.z);
        pack_pair(dst + (cg * 4 + 3) * KPAD2, r0.w, r1.w);
      }
      __syncthreads();
      v16h a0 = load_a_lds(ActS,               HCHUNK, kk, lane);
      v16h a1 = load_a_lds(ActS + 16 * HCHUNK, HCHUNK, kk, lane);
#pragma unroll
      for (int j = 0; j < 8; ++j) {
        v16h bd = load_b_lds(DnS, KPAD2, 0, colbase2 + j * 16, lane);
        acc[0][j] = wmma_f16(a0, bd, acc[0][j]);
        acc[1][j] = wmma_f16(a1, bd, acc[1][j]);
      }
    }
  }

  // epilogue: apply per-row combine weight; shared pass stores, experts atomic-add
#pragma unroll
  for (int rt = 0; rt < 2; ++rt) {
#pragma unroll
    for (int j2 = 0; j2 < 8; ++j2) {
#pragma unroll
      for (int j = 0; j < 8; ++j) {
        int M   = rt * 16 + j + ((lane >> 4) << 3);
        int col = colbase2 + j2 * 16 + (lane & 15);
        float val = acc[rt][j2][j] * Wrow[M];
        size_t oidx = (size_t)Trow[M] * DMODEL + col;
        if (shared_pass) out[oidx] = val;
        else             atomicAdd(&out[oidx], val);
      }
    }
  }
}

extern "C" void kernel_launch(void* const* d_in, const int* in_sizes, int n_in,
                              void* d_out, int out_size, void* d_ws, size_t ws_size,
                              hipStream_t stream) {
  const float* xf     = (const float*)d_in[0];
  const float* gate_w = (const float*)d_in[1];
  const float* eg     = (const float*)d_in[2];
  const float* eu     = (const float*)d_in[3];
  const float* ed     = (const float*)d_in[4];
  const float* sg     = (const float*)d_in[5];
  const float* su     = (const float*)d_in[6];
  const float* sd     = (const float*)d_in[7];
  float* out = (float*)d_out;

  char* ws = (char*)d_ws;
  unsigned* cnt   = (unsigned*)ws;                                   // 8 counters
  float*    fP    = (float*)(ws + 64);                               // f[8], P[8]
  int*      tidx  = (int*)(ws + 256);                                // E*N gathered tokens
  float*    tw    = (float*)(ws + 256 + sizeof(int) * NEXP * NTOK);  // E*N weights

  size_t pool = 2 * (size_t)HCHUNK * KPAD * 2;                 // Gs+Us
  size_t dns  = (size_t)DMODEL * KPAD2 * 2;                    // Down slice
  if (dns > pool) pool = dns;
  size_t shmem = (size_t)RBLK * XLD * 2          // Xs
               + (size_t)RBLK * HCHUNK * 2       // ActS
               + pool                            // staging pool (time-shared)
               + RBLK * 4 + RBLK * 4;            // Wrow, Trow
  (void)hipFuncSetAttribute((const void*)moe_ffn_kernel,
                            hipFuncAttributeMaxDynamicSharedMemorySize, (int)shmem);

  hipLaunchKernelGGL(moe_init_kernel, dim3(1), dim3(64), 0, stream, cnt, fP);
  hipLaunchKernelGGL(moe_router_kernel, dim3(NTOK / 8), dim3(256), 0, stream,
                     xf, gate_w, cnt, fP, tidx, tw);
  hipLaunchKernelGGL(moe_aux_kernel, dim3(1), dim3(64), 0, stream,
                     fP, out + (size_t)NTOK * DMODEL);
  // shared expert: plain stores initialize d_out
  hipLaunchKernelGGL(moe_ffn_kernel, dim3(NTOK / RBLK, 1), dim3(256), shmem, stream,
                     xf, sg, su, sd, (const int*)nullptr, (const float*)nullptr,
                     (const unsigned*)nullptr, out, 1);
  // routed experts: atomic accumulation over gathered rows
  hipLaunchKernelGGL(moe_ffn_kernel, dim3(NTOK / RBLK, NEXP), dim3(256), shmem, stream,
                     xf, eg, eu, ed, tidx, tw, cnt, out, 0);
}
